// SHD_FCNet_84705345011801
// MI455X (gfx1250) — matrix operand order repaired and checked by reading, not compiled, not run on hardware
//
#include <hip/hip_runtime.h>
#include <hip/hip_bf16.h>

typedef __bf16 bf16_t;
typedef __attribute__((ext_vector_type(16))) __bf16 v16bf;
typedef __attribute__((ext_vector_type(8)))  __bf16 v8bf;
typedef __attribute__((ext_vector_type(8)))  float  v8f;

#define T_STEPS 250
#define BATCH   128
#define INDIM   700
#define INPAD   704
#define HDIM    1024
#define OUTDIM  20
#define OUTPAD  32
#define MROWS   (T_STEPS*BATCH)   /* 32000 */

#define ALPHA0_C 0.8187307530779818f   /* exp(-1/5)  */
#define BETA0_C  0.9512294245007140f   /* exp(-1/20) */
#define RHO_C    0.9900498337491681f   /* exp(-1/100)*/
#define THETA_C  1.0f
#define ETA_C    5.0f
#define CONST_C  1.0f

// ---------------------------------------------------------------------------
// Transpose x [B, IN, T] f32  ->  xs [T*B, INPAD] bf16 (zero-padded K)
// ---------------------------------------------------------------------------
__global__ __launch_bounds__(256)
void xpose_x(const float* __restrict__ x, bf16_t* __restrict__ xs)
{
    size_t idx = (size_t)blockIdx.x * 256 + threadIdx.x;
    if (idx >= (size_t)MROWS * INPAD) return;
    int col = (int)(idx % INPAD);
    int row = (int)(idx / INPAD);        // row = t*BATCH + b
    int t = row / BATCH;
    int b = row % BATCH;
    float v = 0.f;
    if (col < INDIM) v = x[((size_t)b * INDIM + col) * T_STEPS + t];
    xs[idx] = (bf16_t)v;
}

// ---------------------------------------------------------------------------
// Weight [K, N] f32 row-major -> transposed bf16 [Npad, Kpad] (zero-padded)
// ---------------------------------------------------------------------------
__global__ __launch_bounds__(256)
void conv_wt(const float* __restrict__ w, bf16_t* __restrict__ wt,
             int K, int Kpad, int N, int Npad)
{
    int idx = blockIdx.x * 256 + threadIdx.x;
    if (idx >= Npad * Kpad) return;
    int n = idx / Kpad;
    int k = idx % Kpad;
    float v = (n < N && k < K) ? w[(size_t)k * N + n] : 0.f;
    wt[idx] = (bf16_t)v;
}

// ---------------------------------------------------------------------------
// bf16 WMMA GEMM: C[M,N] f32 = A[M,K] bf16 row-major x Bt[N,K] bf16 (B transposed)
// 8 waves per WG arranged WGM x WGN, each wave owns MT x NT 16x16 tiles.
// K-fragments per documented CDNA5 wave32 layouts:
//   A: lane = M row (lane&15); lanes>=16 take K+8 half; elems 8..15 at K+16
//   B: lane = N col (lane&15); lanes>=16 take K+8 half; elems 8..15 at K+16
// ---------------------------------------------------------------------------
template<int WGM, int WGN, int MT, int NT>
__global__ __launch_bounds__(32*WGM*WGN)
void gemm_bf16(const bf16_t* __restrict__ A, const bf16_t* __restrict__ Bt,
               float* __restrict__ C, int N, int K)
{
    const int lane = threadIdx.x & 31;
    const int wave = threadIdx.x >> 5;
    const int wm = wave % WGM;
    const int wn = wave / WGM;
    const int rowBase = blockIdx.x * (WGM * MT * 16) + wm * MT * 16;
    const int colBase = blockIdx.y * (WGN * NT * 16) + wn * NT * 16;

    const int am = lane & 15;
    const int ak = (lane >> 4) * 8;     // A K-offset of first half
    const int bn = lane & 15;
    const int bk = (lane >> 4) * 8;     // B K-offset of first half

    v8f acc[MT][NT];
#pragma unroll
    for (int i = 0; i < MT; i++)
#pragma unroll
        for (int j = 0; j < NT; j++)
            acc[i][j] = (v8f){0.f,0.f,0.f,0.f,0.f,0.f,0.f,0.f};

    for (int k0 = 0; k0 < K; k0 += 32) {
        union { v16bf v; v8bf h[2]; } afrag[MT], bfrag[NT];
#pragma unroll
        for (int i = 0; i < MT; i++) {
            const bf16_t* p = A + (size_t)(rowBase + i*16 + am) * K + k0 + ak;
            afrag[i].h[0] = *(const v8bf*)(p);
            afrag[i].h[1] = *(const v8bf*)(p + 16);
        }
#pragma unroll
        for (int j = 0; j < NT; j++) {
            const bf16_t* p = Bt + (size_t)(colBase + j*16 + bn) * K + k0 + bk;
            bfrag[j].h[0] = *(const v8bf*)(p);
            bfrag[j].h[1] = *(const v8bf*)(p + 16);
        }
#pragma unroll
        for (int i = 0; i < MT; i++)
#pragma unroll
            for (int j = 0; j < NT; j++)
                acc[i][j] = __builtin_amdgcn_wmma_f32_16x16x32_bf16(
                    false, afrag[i].v, false, bfrag[j].v,
                    (short)0, acc[i][j], false, false);
    }

    const int crow = (lane >> 4) * 8;
    const int ccol = lane & 15;
#pragma unroll
    for (int i = 0; i < MT; i++)
#pragma unroll
        for (int j = 0; j < NT; j++) {
            float* cp = C + (size_t)(rowBase + i*16 + crow) * N + colBase + j*16 + ccol;
#pragma unroll
            for (int r = 0; r < 8; r++)
                cp[(size_t)r * N] = acc[i][j][r];
        }
}

// ---------------------------------------------------------------------------
// BatchNorm over flattened (T*B): zero accumulators, partial sums, finalize
// ---------------------------------------------------------------------------
__global__ void bn_zero(float* __restrict__ sum, float* __restrict__ sq)
{
    int i = blockIdx.x * 256 + threadIdx.x;
    if (i < HDIM) { sum[i] = 0.f; sq[i] = 0.f; }
}

__global__ __launch_bounds__(256)
void bn_stats(const float* __restrict__ y, float* __restrict__ sum,
              float* __restrict__ sq)
{
    int c  = blockIdx.x * 256 + threadIdx.x;   // column (channel)
    int r0 = blockIdx.y * 256;                 // row chunk
    float s = 0.f, q = 0.f;
    for (int r = 0; r < 256; r++) {
        float v = y[(size_t)(r0 + r) * HDIM + c];
        s += v; q += v * v;
    }
    atomicAdd(&sum[c], s);
    atomicAdd(&sq[c], q);
}

__global__ void bn_finalize(const float* __restrict__ sum, const float* __restrict__ sq,
                            const float* __restrict__ g, const float* __restrict__ bb,
                            float* __restrict__ scale, float* __restrict__ shift)
{
    int c = blockIdx.x * 256 + threadIdx.x;
    if (c >= HDIM) return;
    const float inv = 1.0f / (float)MROWS;
    float mean = sum[c] * inv;
    float var  = sq[c] * inv - mean * mean;
    float sc   = g[c] * rsqrtf(var + 1e-5f);
    scale[c] = sc;
    shift[c] = bb[c] - mean * sc;
}

// ---------------------------------------------------------------------------
// Fused BN-apply + CuAdLIF scan over T; one thread per (b,h) neuron.
// Writes bf16 spikes (exactly 0.0/1.0), GEMM-ready row-major [T*B, H].
// ---------------------------------------------------------------------------
__global__ __launch_bounds__(256)
void adlif_scan_k(const float* __restrict__ y, const float* __restrict__ scale,
                  const float* __restrict__ shift, const float* __restrict__ alpha,
                  const float* __restrict__ beta, const float* __restrict__ a,
                  const float* __restrict__ bco, bf16_t* __restrict__ sout)
{
    int idx = blockIdx.x * 256 + threadIdx.x;   // over B*H
    int h = idx & (HDIM - 1);
    const float sc = scale[h], sh = shift[h];
    const float al = alpha[h], be = beta[h], ac = a[h], bc = bco[h];
    float I = 0.f, v = 0.f, w = 0.f, s = 0.f;
    const size_t stride = (size_t)BATCH * HDIM;
    const float* yp = y + idx;
    bf16_t* sp = sout + idx;
    for (int t = 0; t < T_STEPS; t++) {
        float xt = (*yp) * sc + sh;
        I = al * I + xt;
        w = RHO_C * w + ac * v + bc * s;        // uses previous v, s
        float hm = be * v + I - w;
        s = (hm >= THETA_C) ? 1.f : 0.f;
        v = hm - THETA_C * s;                   // soft reset (detached)
        *sp = (bf16_t)s;
        yp += stride; sp += stride;
    }
}

// ---------------------------------------------------------------------------
// Fused CuLIF scan + spike2time + firing-rate; one thread per (b,o).
// ---------------------------------------------------------------------------
__global__ __launch_bounds__(256)
void out_k(const float* __restrict__ y3, float* __restrict__ out)
{
    int idx = blockIdx.x * 256 + threadIdx.x;
    if (idx >= BATCH * OUTDIM) return;
    int b = idx / OUTDIM, o = idx % OUTDIM;
    const float* yp = y3 + (size_t)b * OUTPAD + o;
    const size_t stride = (size_t)BATCH * OUTPAD;
    float I = 0.f, v = 0.f;
    int   firstT = -1;
    float hFirst = 0.f, hMax = -3.402823466e38f, cnt = 0.f;
    for (int t = 0; t < T_STEPS; t++) {
        float xt = *yp;
        I = ALPHA0_C * I + xt;
        float hm = BETA0_C * v + I;
        float s = (hm >= THETA_C) ? 1.f : 0.f;
        v = hm - THETA_C * s;
        if (s > 0.f) { cnt += 1.f; if (firstT < 0) { firstT = t; hFirst = hm; } }
        hMax = fmaxf(hMax, hm);
        yp += stride;
    }
    float tval;
    if (firstT >= 0) tval = (float)firstT + (THETA_C - hFirst) / THETA_C;
    else             tval = (float)T_STEPS + ETA_C * (THETA_C - hMax) / CONST_C;
    out[idx] = tval;                               // output_times
    out[BATCH * OUTDIM + idx] = cnt / (float)T_STEPS;   // firing_rate
}

// ---------------------------------------------------------------------------
extern "C" void kernel_launch(void* const* d_in, const int* in_sizes, int n_in,
                              void* d_out, int out_size, void* d_ws, size_t ws_size,
                              hipStream_t stream)
{
    const float* x    = (const float*)d_in[0];
    const float* w1   = (const float*)d_in[1];
    const float* bn1g = (const float*)d_in[2];
    const float* bn1b = (const float*)d_in[3];
    const float* al1  = (const float*)d_in[4];
    const float* be1  = (const float*)d_in[5];
    const float* a1   = (const float*)d_in[6];
    const float* b1   = (const float*)d_in[7];
    const float* w2   = (const float*)d_in[8];
    const float* bn2g = (const float*)d_in[9];
    const float* bn2b = (const float*)d_in[10];
    const float* al2  = (const float*)d_in[11];
    const float* be2  = (const float*)d_in[12];
    const float* a2   = (const float*)d_in[13];
    const float* b2   = (const float*)d_in[14];
    const float* w3   = (const float*)d_in[15];
    (void)in_sizes; (void)n_in; (void)out_size; (void)ws_size;

    char* ws = (char*)d_ws;
    size_t off = 0;
    auto alloc = [&](size_t bytes) -> char* {
        char* p = ws + off;
        off += (bytes + 255) & ~(size_t)255;
        return p;
    };
    bf16_t* xs   = (bf16_t*)alloc((size_t)MROWS * INPAD * 2);   // 45 MB
    bf16_t* w1t  = (bf16_t*)alloc((size_t)HDIM * INPAD * 2);
    bf16_t* w2t  = (bf16_t*)alloc((size_t)HDIM * HDIM * 2);
    bf16_t* w3t  = (bf16_t*)alloc((size_t)OUTPAD * HDIM * 2);
    float*  y    = (float*)alloc((size_t)MROWS * HDIM * 4);     // 131 MB (y1/y2)
    bf16_t* s    = (bf16_t*)alloc((size_t)MROWS * HDIM * 2);    // 65 MB (s1/s2)
    float* bnsum   = (float*)alloc(HDIM * 4);
    float* bnsq    = (float*)alloc(HDIM * 4);
    float* bnscale = (float*)alloc(HDIM * 4);
    float* bnshift = (float*)alloc(HDIM * 4);
    float* y3 = (float*)xs;   // xs dead after GEMM1; reuse for y3 [MROWS, OUTPAD]

    const dim3 blk(256);

    // ---- prep: transpose/convert inputs and weights to bf16 ----
    {
        size_t n = (size_t)MROWS * INPAD;
        xpose_x<<<dim3((unsigned)((n + 255) / 256)), blk, 0, stream>>>(x, xs);
    }
    conv_wt<<<dim3((HDIM * INPAD + 255) / 256), blk, 0, stream>>>(w1, w1t, INDIM, INPAD, HDIM, HDIM);
    conv_wt<<<dim3((HDIM * HDIM + 255) / 256), blk, 0, stream>>>(w2, w2t, HDIM, HDIM, HDIM, HDIM);
    conv_wt<<<dim3((OUTPAD * HDIM + 255) / 256), blk, 0, stream>>>(w3, w3t, HDIM, HDIM, OUTDIM, OUTPAD);

    // ---- layer 1: GEMM -> BN stats -> fused BN+AdLIF scan ----
    gemm_bf16<2,4,4,2><<<dim3(MROWS/128, HDIM/128), dim3(256), 0, stream>>>(xs, w1t, y, HDIM, INPAD);
    bn_zero<<<dim3(4), blk, 0, stream>>>(bnsum, bnsq);
    bn_stats<<<dim3(4, MROWS/256), blk, 0, stream>>>(y, bnsum, bnsq);
    bn_finalize<<<dim3(4), blk, 0, stream>>>(bnsum, bnsq, bn1g, bn1b, bnscale, bnshift);
    adlif_scan_k<<<dim3(BATCH*HDIM/256), blk, 0, stream>>>(y, bnscale, bnshift, al1, be1, a1, b1, s);

    // ---- layer 2 ----
    gemm_bf16<2,4,4,2><<<dim3(MROWS/128, HDIM/128), dim3(256), 0, stream>>>(s, w2t, y, HDIM, HDIM);
    bn_zero<<<dim3(4), blk, 0, stream>>>(bnsum, bnsq);
    bn_stats<<<dim3(4, MROWS/256), blk, 0, stream>>>(y, bnsum, bnsq);
    bn_finalize<<<dim3(4), blk, 0, stream>>>(bnsum, bnsq, bn2g, bn2b, bnscale, bnshift);
    adlif_scan_k<<<dim3(BATCH*HDIM/256), blk, 0, stream>>>(y, bnscale, bnshift, al2, be2, a2, b2, s);

    // ---- output layer: narrow GEMM -> fused CuLIF + spike2time + rate ----
    gemm_bf16<8,1,2,2><<<dim3(MROWS/256, 1), dim3(256), 0, stream>>>(s, w3t, y3, OUTPAD, HDIM);
    out_k<<<dim3((BATCH*OUTDIM + 255) / 256), blk, 0, stream>>>(y3, (float*)d_out);
}
